// SelfAttention_7507602833702
// MI455X (gfx1250) — matrix operand order repaired and checked by reading, not compiled, not run on hardware
//
#include <hip/hip_runtime.h>
#include <hip/hip_bf16.h>

// ---------------------------------------------------------------------------
// MI455X (gfx1250) self-attention, bf16 WMMA (v_wmma_f32_16x16x32_bf16).
//
// Two-pass attention: P = dropout(softmax(q k^T / sqrt(512))) materialized in
// bf16 (64 MB -- L2-resident on MI455X's 192 MB L2), then out = P @ v as a
// plain NT GEMM (v stored transposed).  GEMM tiles stream into LDS through
// gfx1250's direct async path (global_load_async_to_lds_b128 / ASYNCcnt /
// s_wait_asynccnt) with double buffering and an L2 prefetch one stage deeper;
// steady-state loop is branch-free: issue(k+1) -> wait(<=4) -> barrier ->
// 8x WMMA -> barrier.
// ---------------------------------------------------------------------------

typedef __bf16 bf16;
typedef __bf16 v16bf __attribute__((ext_vector_type(16)));
typedef __bf16 v8bf  __attribute__((ext_vector_type(8)));
typedef float  v8f   __attribute__((ext_vector_type(8)));

union Frag {
    v16bf v;
    v8bf  h[2];
};

#define BM 128
#define BN 128
#define BK 32

// LDS byte offset = low 32 bits of a generic pointer in the LDS aperture
// (ISA 10.2).
__device__ __forceinline__ unsigned lds_off(const void* p) {
    return (unsigned)(unsigned long long)p;
}
// Async global->LDS 16B copy (wave-wide, per-lane addresses), ASYNCcnt path.
__device__ __forceinline__ void async_ld_b128(unsigned lds_dst, const void* gsrc) {
    asm volatile("global_load_async_to_lds_b128 %0, %1, off"
                 :: "v"(lds_dst), "v"(gsrc) : "memory");
}

// ---------------------------------------------------------------- conversion
__global__ void cvt_f32_bf16(const float* __restrict__ in,
                             bf16* __restrict__ out, long n) {
    long i = (long)blockIdx.x * blockDim.x + threadIdx.x;
    long stride = (long)gridDim.x * blockDim.x;
    for (; i < n; i += stride) out[i] = (bf16)in[i];
}

// ------------------------------------------------------------- generic GEMM
// C[m,n] = sum_k A[m,k] * B[n,k]   (both row-major, K contiguous -> "NT")
// Output: Cf!=null -> f32 row-major; else bf16 (transC: C[n,m] else C[m,n]).
__global__ __launch_bounds__(256) void gemm_nt_bf16(
    const bf16* __restrict__ A, long lda, long strideA,
    const bf16* __restrict__ B, long ldb, long strideB,
    bf16* __restrict__ Cb, float* __restrict__ Cf,
    long ldc, long strideC, int transC, int K)
{
    __shared__ __align__(16) bf16 sA[2][BM * BK];   // 2 x 8 KB
    __shared__ __align__(16) bf16 sB[2][BN * BK];   // 2 x 8 KB

    const int tid  = threadIdx.x;
    const int wave = tid >> 5;
    const int lane = tid & 31;
    const int lh   = lane >> 4;   // lane half (0/1) -> K-half select
    const int ll   = lane & 15;   // row/col within 16

    const long z     = blockIdx.z;
    const long mBase = (long)blockIdx.y * BM;
    const long nBase = (long)blockIdx.x * BN;

    const bf16* Ab = A + z * strideA;
    const bf16* Bb = B + z * strideB;

    const int waveRow = wave & 3;   // 4 wave rows x 32 M
    const int waveCol = wave >> 2;  // 2 wave cols x 64 N
    const int wM = waveRow * 32;
    const int wN = waveCol * 64;

    // Per-thread staging coords: 2 x 16 B chunks per matrix per tile.
    const int lin0 = tid * 2,   lin1 = tid * 2 + 1;
    const int row0 = lin0 >> 2, kp0  = (lin0 & 3) * 8;
    const int row1 = lin1 >> 2, kp1  = (lin1 & 3) * 8;
    // Strength-reduced source pointers: advance by BK elements per tile.
    const bf16* pa0 = &Ab[(mBase + row0) * lda + kp0];
    const bf16* pa1 = &Ab[(mBase + row1) * lda + kp1];
    const bf16* pb0 = &Bb[(nBase + row0) * ldb + kp0];
    const bf16* pb1 = &Bb[(nBase + row1) * ldb + kp1];
    const unsigned la0 = lds_off(&sA[0][row0 * BK + kp0]);
    const unsigned la1 = lds_off(&sA[0][row1 * BK + kp1]);
    const unsigned lb0 = lds_off(&sB[0][row0 * BK + kp0]);
    const unsigned lb1 = lds_off(&sB[0][row1 * BK + kp1]);
    const unsigned bufStride = (unsigned)(BM * BK * sizeof(bf16));  // 8 KB

    auto issue = [&](int buf) {                 // 4 async instrs per tile
        const unsigned o = buf ? bufStride : 0u;
        async_ld_b128(la0 + o, pa0);
        async_ld_b128(la1 + o, pa1);
        async_ld_b128(lb0 + o, pb0);
        async_ld_b128(lb1 + o, pb1);
        pa0 += BK; pa1 += BK; pb0 += BK; pb1 += BK;
    };

    v8f acc[2][4] = {};

    auto compute = [&](int buf) {
        // Fragment loads per ISA 7.12.2 (16-bit A 16x32 / B 32x16, wave32).
        Frag a[2], b[4];
        #pragma unroll
        for (int i = 0; i < 2; ++i) {
            int r = wM + i * 16 + ll;           // A row = lane%16
            a[i].h[0] = *(const v8bf*)&sA[buf][r * BK + lh * 8];
            a[i].h[1] = *(const v8bf*)&sA[buf][r * BK + lh * 8 + 16];
        }
        #pragma unroll
        for (int j = 0; j < 4; ++j) {
            int n = wN + j * 16 + ll;           // B col = lane%16
            b[j].h[0] = *(const v8bf*)&sB[buf][n * BK + lh * 8];
            b[j].h[1] = *(const v8bf*)&sB[buf][n * BK + lh * 8 + 16];
        }
        #pragma unroll
        for (int i = 0; i < 2; ++i)
            #pragma unroll
            for (int j = 0; j < 4; ++j)
                acc[i][j] = __builtin_amdgcn_wmma_f32_16x16x32_bf16(
                    false, a[i].v, false, b[j].v, (short)0, acc[i][j],
                    false, false);
    };

    const int nK = K / BK;
    issue(0);
    for (int kk = 0; kk < nK - 1; ++kk) {       // branch-free steady state
        issue((kk + 1) & 1);
        __builtin_prefetch(pa0, 0, 1);          // warm L2 for tile kk+2
        __builtin_prefetch(pb0, 0, 1);          // (speculative: tail-safe)
        // In-order completion: <=4 outstanding => tile kk has landed.
        asm volatile("s_wait_asynccnt 0x4" ::: "memory");
        __syncthreads();                        // all waves' tile kk visible
        compute(kk & 1);
        __syncthreads();                        // WAR: buffer reused in 2 steps
    }
    asm volatile("s_wait_asynccnt 0x0" ::: "memory");
    __syncthreads();
    compute((nK - 1) & 1);

    // Epilogue: D tile layout -> VGPR r, lane l: M = r + 8*(l/16), N = l%16.
    #pragma unroll
    for (int i = 0; i < 2; ++i)
        #pragma unroll
        for (int j = 0; j < 4; ++j)
            #pragma unroll
            for (int r = 0; r < 8; ++r) {
                long m = mBase + wM + i * 16 + r + 8 * lh;
                long n = nBase + wN + j * 16 + ll;
                float val = acc[i][j][r];
                if (Cf)          Cf[z * strideC + m * ldc + n] = val;
                else if (transC) Cb[z * strideC + n * ldc + m] = (bf16)val;
                else             Cb[z * strideC + m * ldc + n] = (bf16)val;
            }
}

// ----------------------------------------- scores + softmax + dropout -> P
#define TB   16     // q rows per workgroup
#define SB   128    // key-chunk width
#define ATTK 512
#define TT   2048

__global__ __launch_bounds__(256) void attn_scores_softmax(
    const bf16* __restrict__ q, const bf16* __restrict__ k,
    const unsigned char* __restrict__ mask,   // jnp bool_: 1 byte / element
    bf16* __restrict__ P)
{
    extern __shared__ __align__(16) char smemRaw[];
    bf16*  sQ   = (bf16*)smemRaw;             // TB*ATTK bf16     = 16 KB
    bf16*  sK   = sQ + TB * ATTK;             // 2 x SB*BK bf16   = 16 KB
    float* sS   = (float*)(sK + 2 * SB * BK); // TB*TT  f32       = 128 KB
    float* sRed = sS + TB * TT;               // TB*16  f32       =  1 KB
    __shared__ float rowMax[TB], rowSum[TB];

    const int  tid  = threadIdx.x;
    const int  wave = tid >> 5;
    const int  lane = tid & 31;
    const int  lh   = lane >> 4;
    const int  ll   = lane & 15;
    const long b    = blockIdx.y;
    const long t0   = (long)blockIdx.x * TB;

    const bf16* qB = q + (b * TT + t0) * ATTK;
    const bf16* kB = k + b * TT * ATTK;

    // Cache the whole q block (16x512 bf16) in LDS once (async path too).
    #pragma unroll
    for (int c = 0; c < (TB * ATTK) / (8 * 256); ++c) {
        int lin = tid + 256 * c;             // 16B-chunk id
        int row = lin / (ATTK / 8);
        int off = (lin % (ATTK / 8)) * 8;
        async_ld_b128(lds_off(&sQ[row * ATTK + off]), &qB[row * ATTK + off]);
    }

    // k-tile issue: 2 wave-wide async instructions per tile.
    auto issueK = [&](int sc, int kk, int buf) {
        #pragma unroll
        for (int c = 0; c < 2; ++c) {
            int lin = tid * 2 + c;
            int row = lin >> 2;
            int kp  = (lin & 3) * 8;
            async_ld_b128(lds_off(&sK[buf * SB * BK + row * BK + kp]),
                          &kB[(long)(sc * SB + row) * ATTK + kk * BK + kp]);
        }
    };

    const float scale = 0.044194173824159216f;   // 1/sqrt(512)
    const int   nIt   = (TT / SB) * (ATTK / BK); // 16 s-chunks x 16 k-steps

    v8f acc = {};
    auto computeS = [&](int it, int buf) {
        const int kk = it & 15;
        Frag a, bb;
        a.h[0] = *(const v8bf*)&sQ[ll * ATTK + kk * BK + lh * 8];
        a.h[1] = *(const v8bf*)&sQ[ll * ATTK + kk * BK + lh * 8 + 16];
        const int n = wave * 16 + ll;
        bb.h[0] = *(const v8bf*)&sK[buf * SB * BK + n * BK + lh * 8];
        bb.h[1] = *(const v8bf*)&sK[buf * SB * BK + n * BK + lh * 8 + 16];
        acc = __builtin_amdgcn_wmma_f32_16x16x32_bf16(
            false, a.v, false, bb.v, (short)0, acc, false, false);
    };
    auto spill = [&](int sc) {               // s-chunk done -> scaled S to LDS
        #pragma unroll
        for (int r = 0; r < 8; ++r)
            sS[(r + 8 * lh) * TT + sc * SB + wave * 16 + ll] = acc[r] * scale;
        acc = {};
    };

    issueK(0, 0, 0);
    for (int it = 0; it < nIt - 1; ++it) {
        issueK((it + 1) >> 4, (it + 1) & 15, (it + 1) & 1);
        // In-order: <=2 outstanding => sQ block and k-tile `it` have landed.
        asm volatile("s_wait_asynccnt 0x2" ::: "memory");
        __syncthreads();
        computeS(it, it & 1);
        __syncthreads();                     // WAR on sK double buffer
        if ((it & 15) == 15) spill(it >> 4);
    }
    asm volatile("s_wait_asynccnt 0x0" ::: "memory");
    __syncthreads();
    computeS(nIt - 1, (nIt - 1) & 1);
    spill((nIt - 1) >> 4);
    __syncthreads();

    // Row softmax: 16 threads per row over 2048 entries.
    const int row = tid >> 4;
    const int ts  = tid & 15;
    float m = -3.4e38f;
    for (int s = ts; s < TT; s += 16) m = fmaxf(m, sS[row * TT + s]);
    sRed[row * 16 + ts] = m;
    __syncthreads();
    if (ts == 0) {
        float mm = sRed[row * 16];
        for (int i = 1; i < 16; ++i) mm = fmaxf(mm, sRed[row * 16 + i]);
        rowMax[row] = mm;
    }
    __syncthreads();
    const float rm = rowMax[row];
    float sum = 0.f;
    for (int s = ts; s < TT; s += 16) {
        float e = __expf(sS[row * TT + s] - rm);
        sS[row * TT + s] = e;
        sum += e;
    }
    sRed[row * 16 + ts] = sum;
    __syncthreads();
    if (ts == 0) {
        float ss = 0.f;
        for (int i = 0; i < 16; ++i) ss += sRed[row * 16 + i];
        rowSum[row] = ss;
    }
    __syncthreads();

    // P = mask ? exp/sum * 1/(1-p) : 0, stored bf16 (coalesced).
    const float inv_keep = 1.0f / 0.8f;
    const unsigned char* mB = mask + (b * TT + t0) * TT;
    bf16* pB = P + (b * TT + t0) * TT;
    for (int e = tid; e < TB * TT; e += 256) {
        int r = e >> 11;              // / 2048
        int c = e & (TT - 1);
        float pv = mB[(long)r * TT + c]
                 ? sS[e] * (inv_keep / rowSum[r]) : 0.0f;
        pB[(long)r * TT + c] = (bf16)pv;
    }
}

// ------------------------------------------------------------------- launch
extern "C" void kernel_launch(void* const* d_in, const int* in_sizes, int n_in,
                              void* d_out, int out_size, void* d_ws, size_t ws_size,
                              hipStream_t stream)
{
    (void)in_sizes; (void)n_in; (void)out_size; (void)ws_size;

    const float* x  = (const float*)d_in[0];
    const float* Wq = (const float*)d_in[1];
    const float* Wk = (const float*)d_in[2];
    const float* Wv = (const float*)d_in[3];
    const unsigned char* mask = (const unsigned char*)d_in[4];
    float* out = (float*)d_out;

    const long Bn = 8, T = 2048, F = 1024, A = 512;
    const long M  = Bn * T;  // 16384

    // Workspace carve-up (bf16 elements), ~164 MB total.
    bf16* ws  = (bf16*)d_ws;
    bf16* xb  = ws;  ws += M * F;        // x   bf16   32 MB
    bf16* wqb = ws;  ws += A * F;        // Wq  bf16    1 MB
    bf16* wkb = ws;  ws += A * F;        // Wk  bf16    1 MB
    bf16* wvb = ws;  ws += F * F;        // Wv  bf16    2 MB
    bf16* qb  = ws;  ws += M * A;        // q          16 MB
    bf16* kb  = ws;  ws += M * A;        // k          16 MB
    bf16* vT  = ws;  ws += M * F;        // v^T [F,M]  32 MB
    bf16* P   = ws;  ws += Bn * T * T;   // probs      64 MB (L2-resident)

    cvt_f32_bf16<<<4096, 256, 0, stream>>>(x,  xb,  M * F);
    cvt_f32_bf16<<<512,  256, 0, stream>>>(Wq, wqb, A * F);
    cvt_f32_bf16<<<512,  256, 0, stream>>>(Wk, wkb, A * F);
    cvt_f32_bf16<<<1024, 256, 0, stream>>>(Wv, wvb, F * F);

    // q = x Wq^T, k = x Wk^T  (NT: both row-major, K=1024 contiguous)
    gemm_nt_bf16<<<dim3(A / 128, M / 128, 1), 256, 0, stream>>>(
        xb, F, 0, wqb, F, 0, qb, nullptr, A, 0, 0, (int)F);
    gemm_nt_bf16<<<dim3(A / 128, M / 128, 1), 256, 0, stream>>>(
        xb, F, 0, wkb, F, 0, kb, nullptr, A, 0, 0, (int)F);
    // v stored transposed: vT[g, b*T+t] so PV is also an NT GEMM.
    gemm_nt_bf16<<<dim3(F / 128, M / 128, 1), 256, 0, stream>>>(
        xb, F, 0, wvb, F, 0, vT, nullptr, M, 0, 1, (int)F);

    // P = dropout(softmax(q k^T * 1/sqrt(512)))
    size_t smem = (size_t)(TB * ATTK * 2 + 2 * SB * BK * 2
                           + TB * TT * 4 + TB * 16 * 4);   // ~161 KB of 320 KB
    attn_scores_softmax<<<dim3(T / TB, Bn), 256, smem, stream>>>(qb, kb, mask, P);

    // out[b] = P[b] @ v[b]  ==  P[b] (vT[:, b*T : b*T+T])^T, f32 to d_out.
    gemm_nt_bf16<<<dim3(F / 128, T / 128, Bn), 256, 0, stream>>>(
        P, T, T * T, vT, M, T, nullptr, out, F, T * F, 0, (int)T);
}
